// QAGNN_Message_Passing_36249523978268
// MI455X (gfx1250) — matrix operand order repaired and checked by reading, not compiled, not run on hardware
//
#include <hip/hip_runtime.h>
#include <hip/hip_bf16.h>
#include <math.h>

#define HID    256
#define NTYPE  4
#define NEDG   38
#define HEADS  4
#define DPH    64
#define KL     2
#define BB     128
#define NNN    200
#define NNODE  (BB*NNN)          // 25600
#define E_REAL 150000
#define ET     (E_REAL + NNODE)  // 175600
#define EPSBN  1e-5f

typedef _Float16 v16h __attribute__((ext_vector_type(16)));
typedef _Float16 v8h  __attribute__((ext_vector_type(8)));
typedef _Float16 v4h  __attribute__((ext_vector_type(4)));
typedef float    v8f  __attribute__((ext_vector_type(8)));
typedef float    v4f  __attribute__((ext_vector_type(4)));

#if __has_builtin(__builtin_amdgcn_global_load_async_to_lds_b128) && \
    __has_builtin(__builtin_amdgcn_s_wait_asynccnt)
#define HAVE_ASYNC_LDS 1
// Builtin signature (from hipcc diagnostic): params are int4 pointers in
// addrspace(1) (global) and addrspace(3) (LDS).
typedef int vi4 __attribute__((vector_size(16)));
typedef __attribute__((address_space(1))) vi4 gvi4;
typedef __attribute__((address_space(3))) vi4 lvi4;
#endif

__device__ __forceinline__ float gelu_f(float x) {
    return 0.5f * x * (1.0f + erff(x * 0.70710678118654752f));
}

__device__ __forceinline__ void atomicMaxF(float* addr, float val) {
    int* ia = (int*)addr;
    int cur = *ia;
    while (__int_as_float(cur) < val) {
        int prev = atomicCAS(ia, cur, __float_as_int(val));
        if (prev == cur) break;
        cur = prev;
    }
}

// ---------------------------------------------------------------------------
// Pre-convert weight W[Kd,256] f32 (row-major) -> Wh[256][Kd] f16 (transposed)
// so the GEMM B-stage is a raw b128 global->LDS copy.
// ---------------------------------------------------------------------------
__global__ void k_wconvert(const float* __restrict__ W, int Kd,
                           _Float16* __restrict__ Wh) {
    int idx = blockIdx.x * blockDim.x + threadIdx.x;
    if (idx >= 256 * Kd) return;
    int k = idx >> 8, n = idx & 255;                 // coalesced read of W row k
    Wh[(size_t)n * Kd + k] = (_Float16)W[(size_t)k * 256 + n];
}

// ---------------------------------------------------------------------------
// Tiled WMMA GEMM: C[M,256] = epi(A[M,Kd] @ W[Kd,256] + bias (+Cadd))
// Block = 256 threads = 8 waves, tile 128x128, wave owns 32x64 (2x4 WMMA accs).
// A staged f32->f16 with clamped (branchless) b128 loads; B tile copied from
// pre-converted transposed f16 weights (async-to-LDS when available).
// epi: 0 = none, 1 = relu(bn(.)), 2 = gelu(.)
// ---------------------------------------------------------------------------
__global__ __launch_bounds__(256) void gemm_wmma(
    const float* __restrict__ A, int M, int Kd, int lda,
    const _Float16* __restrict__ Wh,                 // [256][Kd] transposed f16
    const float* __restrict__ bias,
    const float* __restrict__ bng, const float* __restrict__ bnb,
    const float* __restrict__ Cadd,
    float* __restrict__ C, int ldc, int epi)
{
    __shared__ __align__(16) _Float16 Ah[128 * 32];   // row-major, stride 32
    __shared__ __align__(16) _Float16 Bt[128 * 32];   // col-major: Bt[n*32+k]

    const int tid  = threadIdx.x;
    const int lane = tid & 31;
    const int w    = tid >> 5;
    const int wm   = (w & 3) * 32;
    const int wn   = (w >> 2) * 64;
    const int lrow = lane & 15;
    const int hi   = lane >> 4;
    const int tileN = blockIdx.x * 128;
    const int tileM = blockIdx.y * 128;

    v8f acc[2][4] = {};

    for (int k0 = 0; k0 < Kd; k0 += 32) {
        // --- B tile: pure 16B copies from transposed f16 weights ---
#ifdef HAVE_ASYNC_LDS
        {
            int i0 = tid;           // v8h chunk ids
            int i1 = tid + 256;
            int n0 = i0 >> 2, c0 = i0 & 3;
            int n1 = i1 >> 2, c1 = i1 & 3;
            __builtin_amdgcn_global_load_async_to_lds_b128(
                (gvi4*)&Wh[(size_t)(tileN + n0) * Kd + k0 + c0 * 8],
                (lvi4*)&Bt[n0 * 32 + c0 * 8], 0, 0);
            __builtin_amdgcn_global_load_async_to_lds_b128(
                (gvi4*)&Wh[(size_t)(tileN + n1) * Kd + k0 + c1 * 8],
                (lvi4*)&Bt[n1 * 32 + c1 * 8], 0, 0);
        }
#else
        #pragma unroll
        for (int it = 0; it < 2; ++it) {
            int i = tid + it * 256;
            int n = i >> 2, c8 = i & 3;
            *(v8h*)&Bt[n * 32 + c8 * 8] =
                *(const v8h*)&Wh[(size_t)(tileN + n) * Kd + k0 + c8 * 8];
        }
#endif
        // --- A tile: branchless clamped b128 loads, convert, b64 LDS store ---
        #pragma unroll
        for (int it = 0; it < 4; ++it) {
            int i = tid + it * 256;          // float4 id in [0,1024)
            int r = i >> 3, c4 = i & 7;
            int gr = tileM + r;
            if (gr >= M) gr = M - 1;         // clamp: garbage rows never stored
            v4f v = *(const v4f*)&A[(size_t)gr * lda + k0 + c4 * 4];
            *(v4h*)&Ah[r * 32 + c4 * 4] = __builtin_convertvector(v, v4h);
        }
#ifdef HAVE_ASYNC_LDS
        __builtin_amdgcn_s_wait_asynccnt(0);
#endif
        __syncthreads();

        if (k0 + 32 < Kd) {
            __builtin_prefetch(&A[(size_t)(tileM + (tid >> 1)) * lda + k0 + 32], 0, 1);
        }

        v16h afrag[2];
        #pragma unroll
        for (int r = 0; r < 2; ++r) {
            int row = wm + r * 16 + lrow;
            v8h lo = *(const v8h*)&Ah[row * 32 + hi * 8];
            v8h hh = *(const v8h*)&Ah[row * 32 + 16 + hi * 8];
            afrag[r] = __builtin_shufflevector(lo, hh,
                0,1,2,3,4,5,6,7,8,9,10,11,12,13,14,15);
        }
        v16h bfrag[4];
        #pragma unroll
        for (int c = 0; c < 4; ++c) {
            int col = wn + c * 16 + lrow;
            v8h lo = *(const v8h*)&Bt[col * 32 + hi * 8];
            v8h hh = *(const v8h*)&Bt[col * 32 + 16 + hi * 8];
            bfrag[c] = __builtin_shufflevector(lo, hh,
                0,1,2,3,4,5,6,7,8,9,10,11,12,13,14,15);
        }
        #pragma unroll
        for (int r = 0; r < 2; ++r)
            #pragma unroll
            for (int c = 0; c < 4; ++c)
                acc[r][c] = __builtin_amdgcn_wmma_f32_16x16x32_f16(
                    false, afrag[r], false, bfrag[c],
                    (short)0, acc[r][c], false, false);
        __syncthreads();
    }

    const float bnscale = rsqrtf(1.0f + EPSBN);
    #pragma unroll
    for (int r = 0; r < 2; ++r) {
        #pragma unroll
        for (int c = 0; c < 4; ++c) {
            int col   = tileN + wn + c * 16 + lrow;
            float bco = bias ? bias[col] : 0.0f;
            float gco = (epi == 1) ? bng[col] * bnscale : 0.0f;
            float eco = (epi == 1) ? bnb[col] : 0.0f;
            #pragma unroll
            for (int i = 0; i < 8; ++i) {
                int row = tileM + wm + r * 16 + hi * 8 + i;
                if (row < M) {
                    float v = acc[r][c][i] + bco;
                    if (Cadd) v += Cadd[(size_t)row * ldc + col];
                    if (epi == 1)      v = fmaxf(v * gco + eco, 0.0f);
                    else if (epi == 2) v = gelu_f(v);
                    C[(size_t)row * ldc + col] = v;
                }
            }
        }
    }
}

// ---------------------------------------------------------------------------
// Helper kernels
// ---------------------------------------------------------------------------
__global__ void k_fill(float* p, float v, long n) {
    long i = (long)blockIdx.x * blockDim.x + threadIdx.x;
    long stride = (long)gridDim.x * blockDim.x;
    for (; i < n; i += stride) p[i] = v;
}

__global__ void k_count(const int* __restrict__ ei, float* __restrict__ cnt) {
    int e = blockIdx.x * blockDim.x + threadIdx.x;
    if (e >= ET) return;
    int s = (e < E_REAL) ? ei[e] : (e - E_REAL);
    atomicAdd(&cnt[s], 1.0f);
}

__global__ void k_node_extra(const float* __restrict__ ns, const int* __restrict__ ntv,
                             const float* __restrict__ Wnt, const float* __restrict__ bnt,
                             const float* __restrict__ Wsc, const float* __restrict__ bsc,
                             float* __restrict__ extra) {
    int n = blockIdx.x;
    int j = threadIdx.x; // 128
    __shared__ float sinb[128];
    float sc = ns[n];
    sinb[j] = sinf(powf(1.1f, (float)j) * sc);
    __syncthreads();
    int t = ntv[n];
    extra[(size_t)n * HID + j] = gelu_f(Wnt[t * 128 + j] + bnt[j]);
    float acc = bsc[j];
    #pragma unroll 4
    for (int i = 0; i < 128; ++i) acc += sinb[i] * Wsc[i * 128 + j];
    extra[(size_t)n * HID + 128 + j] = gelu_f(acc);
}

__global__ void k_edge_hidden(const int* __restrict__ ei, const int* __restrict__ etype,
                              const int* __restrict__ ntv,
                              const float* __restrict__ W1, const float* __restrict__ b1,
                              const float* __restrict__ g, const float* __restrict__ be,
                              float* __restrict__ out) {
    int e = blockIdx.x;
    int j = threadIdx.x; // 256
    int rA, nH, nT;
    if (e < E_REAL) { rA = etype[e]; nH = ntv[ei[e]]; nT = ntv[ei[E_REAL + e]]; }
    else { int n = e - E_REAL; rA = NEDG; nH = ntv[n]; nT = nH; }
    float v = W1[rA * HID + j] + W1[(NEDG + 1 + nH) * HID + j]
            + W1[(NEDG + 1 + NTYPE + nT) * HID + j] + b1[j];
    v = v * (g[j] * rsqrtf(1.0f + EPSBN)) + be[j];
    out[(size_t)e * HID + j] = fmaxf(v, 0.0f);
}

__global__ void k_concat(const float* __restrict__ X, const float* __restrict__ extra,
                         float* __restrict__ xcat) {
    long i = (long)blockIdx.x * blockDim.x + threadIdx.x;
    if (i >= (long)NNODE * 512) return;
    int n = (int)(i >> 9), j = (int)(i & 511);
    xcat[i] = (j < 256) ? X[(size_t)n * 256 + j] : extra[(size_t)n * 256 + (j - 256)];
}

__global__ void k_scores(const float* __restrict__ Xq, const float* __restrict__ Xk,
                         const float* __restrict__ ek, const int* __restrict__ ei,
                         float* __restrict__ scores, float* __restrict__ smax) {
    int idx = blockIdx.x * blockDim.x + threadIdx.x;
    if (idx >= ET * HEADS) return;
    int e = idx >> 2, h = idx & 3;
    int s, d;
    if (e < E_REAL) { s = ei[e]; d = ei[E_REAL + e]; } else { s = d = e - E_REAL; }
    const v4f* q  = (const v4f*)(Xq + (size_t)s * HID + h * DPH);
    const v4f* kk = (const v4f*)(Xk + (size_t)d * HID + h * DPH);
    const v4f* ee = (const v4f*)(ek + (size_t)e * HID + h * DPH);
    float acc = 0.0f;
    #pragma unroll
    for (int i = 0; i < DPH / 4; ++i) {
        v4f qv = q[i], kv = kk[i], ev = ee[i];
        acc += qv[0] * (kv[0] + ev[0]) + qv[1] * (kv[1] + ev[1])
             + qv[2] * (kv[2] + ev[2]) + qv[3] * (kv[3] + ev[3]);
    }
    acc *= 0.125f; // 1/sqrt(DPH)
    scores[idx] = acc;
    atomicMaxF(&smax[s * HEADS + h], acc);
}

__global__ void k_expsum(const int* __restrict__ ei, float* __restrict__ scores,
                         const float* __restrict__ smax, float* __restrict__ ssum) {
    int idx = blockIdx.x * blockDim.x + threadIdx.x;
    if (idx >= ET * HEADS) return;
    int e = idx >> 2, h = idx & 3;
    int s = (e < E_REAL) ? ei[e] : (e - E_REAL);
    float al = expf(scores[idx] - smax[s * HEADS + h]);
    scores[idx] = al;
    atomicAdd(&ssum[s * HEADS + h], al);
}

__global__ void k_aggregate(const float* __restrict__ Xm, const float* __restrict__ em,
                            const float* __restrict__ scores, const float* __restrict__ ssum,
                            const float* __restrict__ cnt, const int* __restrict__ ei,
                            float* __restrict__ aggr) {
    int e = blockIdx.x;
    int j = threadIdx.x; // 256
    int s, d;
    if (e < E_REAL) { s = ei[e]; d = ei[E_REAL + e]; } else { s = d = e - E_REAL; }
    int h = j >> 6;
    float al = scores[e * HEADS + h] / ssum[s * HEADS + h] * cnt[s];
    float v = (Xm[(size_t)s * HID + j] + em[(size_t)e * HID + j]) * al;
    atomicAdd(&aggr[(size_t)d * HID + j], v);
}

// ---------------------------------------------------------------------------
static inline void gemm(hipStream_t s, const float* A, int M, int Kd, int lda,
                        const float* W, _Float16* wbuf, const float* bias,
                        const float* g, const float* be, const float* Cadd,
                        float* C, int epi) {
    k_wconvert<<<(256 * Kd + 255) / 256, 256, 0, s>>>(W, Kd, wbuf);
    dim3 grid(HID / 128, (M + 127) / 128);
    gemm_wmma<<<grid, 256, 0, s>>>(A, M, Kd, lda, wbuf, bias, g, be, Cadd, C, HID, epi);
}

extern "C" void kernel_launch(void* const* d_in, const int* in_sizes, int n_in,
                              void* d_out, int out_size, void* d_ws, size_t ws_size,
                              hipStream_t stream) {
    (void)in_sizes; (void)n_in; (void)out_size; (void)ws_size;
    const float* H     = (const float*)d_in[0];
    const float* nsc   = (const float*)d_in[1];
    const float* Wnt   = (const float*)d_in[2];
    const float* bnt   = (const float*)d_in[3];
    const float* Wsc   = (const float*)d_in[4];
    const float* bsc   = (const float*)d_in[5];
    const float* eeW1  = (const float*)d_in[6];
    const float* eeb1  = (const float*)d_in[7];
    const float* eeg   = (const float*)d_in[8];
    const float* eebe  = (const float*)d_in[9];
    const float* eeW2  = (const float*)d_in[10];
    const float* eeb2  = (const float*)d_in[11];
    const float* Wk    = (const float*)d_in[12];
    const float* bk    = (const float*)d_in[13];
    const float* Wm    = (const float*)d_in[14];
    const float* bm    = (const float*)d_in[15];
    const float* Wq    = (const float*)d_in[16];
    const float* bq    = (const float*)d_in[17];
    const float* mW1   = (const float*)d_in[18];
    const float* mb1   = (const float*)d_in[19];
    const float* mg    = (const float*)d_in[20];
    const float* mbe   = (const float*)d_in[21];
    const float* mW2   = (const float*)d_in[22];
    const float* mb2   = (const float*)d_in[23];
    const float* Wvh   = (const float*)d_in[24];
    const float* bvh   = (const float*)d_in[25];
    const float* Wvx   = (const float*)d_in[26];
    const float* bvx   = (const float*)d_in[27];
    const int*   ntv   = (const int*)d_in[28];
    const int*   ei    = (const int*)d_in[29];
    const int*   etype = (const int*)d_in[30];

    float* ws = (float*)d_ws;
    size_t off = 0;
    auto alloc = [&](size_t n) { float* p = ws + off; off += n; return p; };
    float* extra  = alloc((size_t)NNODE * HID);
    float* xcat   = alloc((size_t)NNODE * 512);
    float* Xk     = alloc((size_t)NNODE * HID);
    float* Xm     = alloc((size_t)NNODE * HID);
    float* Xq     = alloc((size_t)NNODE * HID);
    float* aggr   = alloc((size_t)NNODE * HID);
    float* Xbuf   = alloc((size_t)NNODE * HID);
    float* h1buf  = alloc((size_t)NNODE * HID);
    float* tmpC   = alloc((size_t)NNODE * HID);
    float* cnt    = alloc((size_t)NNODE);
    float* smax   = alloc((size_t)NNODE * HEADS);
    float* ssum   = alloc((size_t)NNODE * HEADS);
    float* wbuf_f = alloc((size_t)256 * 512 / 2);      // 256*512 halves
    float* scores = alloc((size_t)ET * HEADS);
    float* eemb   = alloc((size_t)ET * HID);
    float* bufA   = alloc((size_t)ET * HID);           // h_edge -> edge_k -> edge_m
    _Float16* wbuf = (_Float16*)wbuf_f;

    // degree counts over src (includes self loops)
    k_fill<<<256, 256, 0, stream>>>(cnt, 0.0f, (long)NNODE);
    k_count<<<(ET + 255) / 256, 256, 0, stream>>>(ei, cnt);

    // node extra features
    k_node_extra<<<NNODE, 128, 0, stream>>>(nsc, ntv, Wnt, bnt, Wsc, bsc, extra);

    // edge embedding: one-hot matmul via row gathers, then dense WMMA GEMM
    k_edge_hidden<<<ET, HID, 0, stream>>>(ei, etype, ntv, eeW1, eeb1, eeg, eebe, bufA);
    gemm(stream, bufA, ET, HID, HID, eeW2, wbuf, eeb2, nullptr, nullptr, nullptr, eemb, 0);

    for (int l = 0; l < KL; ++l) {
        const float* Xcur = (l == 0) ? H : Xbuf;
        k_concat<<<(int)(((long)NNODE * 512 + 255) / 256), 256, 0, stream>>>(Xcur, extra, xcat);

        const float* Wkl = Wk + (size_t)l * 768 * HID;
        const float* Wml = Wm + (size_t)l * 768 * HID;
        const float* Wql = Wq + (size_t)l * 512 * HID;

        // node-side projections (concat-GEMM factorized into node + edge parts)
        gemm(stream, xcat, NNODE, 512, 512, Wkl, wbuf, nullptr, nullptr, nullptr, nullptr, Xk, 0);
        gemm(stream, xcat, NNODE, 512, 512, Wml, wbuf, nullptr, nullptr, nullptr, nullptr, Xm, 0);
        gemm(stream, xcat, NNODE, 512, 512, Wql, wbuf, bq + (size_t)l * HID, nullptr, nullptr, nullptr, Xq, 0);

        // edge-side key projection
        gemm(stream, eemb, ET, HID, HID, Wkl + (size_t)512 * HID, wbuf,
             bk + (size_t)l * HID, nullptr, nullptr, nullptr, bufA, 0);

        k_fill<<<256, 256, 0, stream>>>(smax, -1e30f, (long)NNODE * HEADS);
        k_fill<<<256, 256, 0, stream>>>(ssum, 0.0f, (long)NNODE * HEADS);
        k_fill<<<256, 256, 0, stream>>>(aggr, 0.0f, (long)NNODE * HID);

        k_scores<<<(ET * HEADS + 255) / 256, 256, 0, stream>>>(Xq, Xk, bufA, ei, scores, smax);
        k_expsum<<<(ET * HEADS + 255) / 256, 256, 0, stream>>>(ei, scores, smax, ssum);

        // edge-side message projection (reuses bufA after scores are done)
        gemm(stream, eemb, ET, HID, HID, Wml + (size_t)512 * HID, wbuf,
             bm + (size_t)l * HID, nullptr, nullptr, nullptr, bufA, 0);

        k_aggregate<<<ET, HID, 0, stream>>>(Xm, bufA, scores, ssum, cnt, ei, aggr);

        // node MLP: relu(bn(aggr@W1+b1)) then gelu(h1@W2+b2)
        gemm(stream, aggr, NNODE, HID, HID, mW1 + (size_t)l * HID * HID, wbuf,
             mb1 + (size_t)l * HID, mg + (size_t)l * HID, mbe + (size_t)l * HID,
             nullptr, h1buf, 1);
        gemm(stream, h1buf, NNODE, HID, HID, mW2 + (size_t)l * HID * HID, wbuf,
             mb2 + (size_t)l * HID, nullptr, nullptr, nullptr, Xbuf, 2);
    }

    // out = gelu(H@Wvh + bvh + X@Wvx + bvx)
    gemm(stream, H, NNODE, HID, HID, Wvh, wbuf, bvh, nullptr, nullptr, nullptr, tmpC, 0);
    gemm(stream, Xbuf, NNODE, HID, HID, Wvx, wbuf, bvx, nullptr, nullptr, tmpC,
         (float*)d_out, 2);
}
